// ChainIterator_1400159339038
// MI455X (gfx1250) — compile-verified, matrix operand order checked by reading, not run on hardware
//
#include <hip/hip_runtime.h>
#include <hip/hip_bf16.h>
#include <math.h>

#define H 128
#define BATCH 16
#define TN 64   // nodes per block in k1 / k4

typedef __attribute__((ext_vector_type(16))) __bf16         v16bf;
typedef __attribute__((ext_vector_type(16))) unsigned short v16u;
typedef __attribute__((ext_vector_type(8)))  float          v8f;

__device__ __forceinline__ unsigned short f2bf(float x) {
  unsigned int u = __float_as_uint(x);
  u += 0x7FFFu + ((u >> 16) & 1u);   // round-to-nearest-even
  return (unsigned short)(u >> 16);
}
__device__ __forceinline__ v16bf as_bf(v16u x) {
  union { v16u u; v16bf b; } c; c.u = x; return c.b;
}

// ---------------------------------------------------------------------------
// k0: hsb = state @ W1[:H] + b1 ; pack W1[H:] into bf16 WMMA B-fragment
//     layout ; zero gsum / context accumulators (fresh each launch).
// ---------------------------------------------------------------------------
__global__ void k0_setup(const float* __restrict__ state,
                         const float* __restrict__ W1,
                         const float* __restrict__ b1,
                         float* __restrict__ hsb,
                         unsigned short* __restrict__ Bpk,
                         float* __restrict__ gsum,
                         float* __restrict__ context) {
  int t = threadIdx.x;
  for (int o = t; o < BATCH * H; o += 256) {
    int b = o >> 7, hc = o & 127;
    float acc = b1[hc];
    for (int k = 0; k < H; ++k)
      acc = fmaf(state[b * H + k], W1[k * H + hc], acc);
    hsb[o] = acc;
  }
  // B-fragment: for (kt,nt), lane L holds column n = nt*16 + L%16,
  // element j -> K = kt*32 + (j<8 ? j : j+8) + 8*(L/16)   (ISA 7.12.2)
  for (int idx = t; idx < 4 * 8 * 32 * 16; idx += 256) {
    int j    = idx & 15;
    int lane = (idx >> 4) & 31;
    int nt   = (idx >> 9) & 7;
    int kt   = idx >> 12;
    int K = kt * 32 + (j < 8 ? j : j + 8) + 8 * (lane >> 4);
    int n = nt * 16 + (lane & 15);
    Bpk[idx] = f2bf(W1[(size_t)(H + K) * H + n]);
  }
  if (t < BATCH) gsum[t] = 0.0f;
  for (int o = t; o < BATCH * H; o += 256) context[o] = 0.0f;
}

// ---------------------------------------------------------------------------
// k1: per 64-node block: hn = nodes @ W1[H:] via bf16 WMMA into LDS, then
//     fused score MLP  score[b,n] = relu(hsb[b]+hn[n]) . W2 + b2,
//     plus per-block per-batch running max for softmax.
// ---------------------------------------------------------------------------
__global__ void __launch_bounds__(256)
k1_scores(const float* __restrict__ nodes,
          const unsigned short* __restrict__ Bpk,
          const float* __restrict__ hsb,
          const float* __restrict__ W2,
          const float* __restrict__ b2,
          float* __restrict__ scores,
          float* __restrict__ pmax,
          int N) {
  __shared__ float hn_s[TN * 129];        // stride 129 -> conflict-free reads
  __shared__ float hsb_s[BATCH * H];
  __shared__ float w2_s[H];
  __shared__ float sc_s[BATCH * TN];

  const int t = threadIdx.x;
  const int node0 = blockIdx.x * TN;

  // hide HBM latency on the one real streaming input (-> global_prefetch_b8)
  if (t < TN) {
    int pn = node0 + t;
    if (pn < N) __builtin_prefetch(nodes + (size_t)pn * H, 0, 1);
  }

  for (int o = t; o < BATCH * H; o += 256) hsb_s[o] = hsb[o];
  if (t < H) w2_s[t] = W2[t];

  const int lane = t & 31;
  const int w    = t >> 5;       // 8 waves
  const int rt   = w & 3;        // row tile (16 nodes)
  const int ng   = w >> 2;       // column-tile group (4 tiles of 16)
  const int m    = lane & 15;
  const int hh   = lane >> 4;
  const int node = node0 + rt * 16 + m;

  v8f acc[4] = {};
  #pragma unroll
  for (int kt = 0; kt < 4; ++kt) {
    // A-fragment: lane holds K = kt*32+8h .. +7  and  kt*32+16+8h .. +7
    v16u au;
    if (node < N) {
      const float* rp = nodes + (size_t)node * H + kt * 32 + 8 * hh;
      float4 q0 = *(const float4*)(rp);
      float4 q1 = *(const float4*)(rp + 4);
      float4 q2 = *(const float4*)(rp + 16);
      float4 q3 = *(const float4*)(rp + 20);
      au[0]  = f2bf(q0.x); au[1]  = f2bf(q0.y); au[2]  = f2bf(q0.z); au[3]  = f2bf(q0.w);
      au[4]  = f2bf(q1.x); au[5]  = f2bf(q1.y); au[6]  = f2bf(q1.z); au[7]  = f2bf(q1.w);
      au[8]  = f2bf(q2.x); au[9]  = f2bf(q2.y); au[10] = f2bf(q2.z); au[11] = f2bf(q2.w);
      au[12] = f2bf(q3.x); au[13] = f2bf(q3.y); au[14] = f2bf(q3.z); au[15] = f2bf(q3.w);
    } else {
      #pragma unroll
      for (int j = 0; j < 16; ++j) au[j] = 0;
    }
    v16bf a = as_bf(au);
    #pragma unroll
    for (int i = 0; i < 4; ++i) {
      int nt = ng * 4 + i;
      v16u bu = *(const v16u*)(Bpk + (((kt * 8 + nt) * 32 + lane) << 4));
      acc[i] = __builtin_amdgcn_wmma_f32_16x16x32_bf16(
          false, a, false, as_bf(bu), (short)0, acc[i], false, false);
    }
  }

  // C/D layout: VGPR r holds M = r + 8*(lane/16), N = lane%16
  #pragma unroll
  for (int i = 0; i < 4; ++i) {
    int n = (ng * 4 + i) * 16 + m;
    #pragma unroll
    for (int r = 0; r < 8; ++r) {
      int mr = rt * 16 + r + 8 * hh;
      hn_s[mr * 129 + n] = acc[i][r];
    }
  }
  __syncthreads();

  // fused score MLP: k outer, 4 batches inner -> each hn/w2 LDS read
  // amortized over 4 FMAs
  const int nl = t & 63;
  const int bq = t >> 6;
  const float bias2 = b2[0];
  const int gnode = node0 + nl;
  {
    const float* np_ = hn_s + nl * 129;
    const float* hp0 = hsb_s + (bq * 4 + 0) * H;
    const float* hp1 = hsb_s + (bq * 4 + 1) * H;
    const float* hp2 = hsb_s + (bq * 4 + 2) * H;
    const float* hp3 = hsb_s + (bq * 4 + 3) * H;
    float s0 = 0.0f, s1 = 0.0f, s2 = 0.0f, s3 = 0.0f;
    #pragma unroll 4
    for (int k = 0; k < H; ++k) {
      float hv = np_[k];
      float wk = w2_s[k];
      float v0 = fmaxf(hp0[k] + hv, 0.0f);
      float v1 = fmaxf(hp1[k] + hv, 0.0f);
      float v2 = fmaxf(hp2[k] + hv, 0.0f);
      float v3 = fmaxf(hp3[k] + hv, 0.0f);
      s0 = fmaf(v0, wk, s0);
      s1 = fmaf(v1, wk, s1);
      s2 = fmaf(v2, wk, s2);
      s3 = fmaf(v3, wk, s3);
    }
    float ss[4] = {s0, s1, s2, s3};
    #pragma unroll
    for (int bi = 0; bi < 4; ++bi) {
      int b = bq * 4 + bi;
      float s = ss[bi] + bias2;
      if (gnode < N) scores[(size_t)b * N + gnode] = s;
      sc_s[b * TN + nl] = (gnode < N) ? s : -INFINITY;
    }
  }
  __syncthreads();
  if (t < BATCH) {
    float mx = -INFINITY;
    for (int i = 0; i < TN; ++i) mx = fmaxf(mx, sc_s[t * TN + i]);
    pmax[blockIdx.x * BATCH + t] = mx;
  }
}

// ---------------------------------------------------------------------------
// k2: reduce per-block maxima -> gmax[16]
// ---------------------------------------------------------------------------
__global__ void k2_gmax(const float* __restrict__ pmax,
                        float* __restrict__ gmax, int nB) {
  __shared__ float red[256];
  int t = threadIdx.x;
  int b = t & 15, ch = t >> 4;
  float m = -INFINITY;
  for (int blk = ch; blk < nB; blk += 16)
    m = fmaxf(m, pmax[blk * BATCH + b]);
  red[t] = m;
  __syncthreads();
  if (t < BATCH) {
    float mm = -INFINITY;
    for (int c = 0; c < 16; ++c) mm = fmaxf(mm, red[c * 16 + t]);
    gmax[t] = mm;
  }
}

// ---------------------------------------------------------------------------
// k3: gsum[b] = sum_n exp(score - gmax)   (scores are L2-resident)
// ---------------------------------------------------------------------------
__global__ void k3_sumexp(const float* __restrict__ scores,
                          const float* __restrict__ gmax,
                          float* __restrict__ gsum,
                          int N, int NC) {
  __shared__ float red[256];
  int t = threadIdx.x;
  int b = blockIdx.x / NC;
  int c = blockIdx.x % NC;
  int base = c * 4096;
  int end = min(base + 4096, N);
  float gm = gmax[b];
  float s = 0.0f;
  for (int idx = base + t; idx < end; idx += 256)
    s += expf(scores[(size_t)b * N + idx] - gm);
  red[t] = s;
  __syncthreads();
  for (int off = 128; off > 0; off >>= 1) {
    if (t < off) red[t] += red[t + off];
    __syncthreads();
  }
  if (t == 0) atomicAdd(&gsum[b], red[0]);
}

// ---------------------------------------------------------------------------
// k4: context[b] += sum_n softmax(score) * nodes[n]   (nodes staged in LDS)
// ---------------------------------------------------------------------------
__global__ void __launch_bounds__(256)
k4_context(const float* __restrict__ nodes,
           const float* __restrict__ scores,
           const float* __restrict__ gmax,
           const float* __restrict__ gsum,
           float* __restrict__ context,
           int N) {
  __shared__ float ns[TN * H];
  __shared__ float wv[BATCH * TN];
  __shared__ float rg[BATCH];
  int t = threadIdx.x;
  int node0 = blockIdx.x * TN;
  if (t < BATCH) rg[t] = 1.0f / gsum[t];
  for (int idx = t; idx < TN * H; idx += 256) {
    int node = node0 + (idx >> 7);
    ns[idx] = (node < N) ? nodes[(size_t)node * H + (idx & 127)] : 0.0f;
  }
  __syncthreads();
  for (int idx = t; idx < BATCH * TN; idx += 256) {
    int b = idx / TN, nl = idx % TN;
    int node = node0 + nl;
    wv[idx] = (node < N)
        ? expf(scores[(size_t)b * N + node] - gmax[b]) * rg[b] : 0.0f;
  }
  __syncthreads();
  int b  = t >> 4;
  int c0 = (t & 15) * 8;
  float acc[8] = {0};
  for (int nl = 0; nl < TN; ++nl) {
    float w = wv[b * TN + nl];
    const float* rp = ns + nl * H + c0;
    #pragma unroll
    for (int i = 0; i < 8; ++i) acc[i] = fmaf(w, rp[i], acc[i]);
  }
  #pragma unroll
  for (int i = 0; i < 8; ++i) atomicAdd(&context[b * H + c0 + i], acc[i]);
}

// ---------------------------------------------------------------------------
// k5: GRUCell(context, state) -> out
// ---------------------------------------------------------------------------
__global__ void k5_gru(const float* __restrict__ context,
                       const float* __restrict__ state,
                       const float* __restrict__ w_ih,
                       const float* __restrict__ w_hh,
                       const float* __restrict__ b_ih,
                       const float* __restrict__ b_hh,
                       float* __restrict__ out) {
  int t = threadIdx.x;
  for (int o = t; o < BATCH * H; o += 256) {
    int b = o >> 7, hc = o & 127;
    const float* cx = context + b * H;
    const float* st = state + b * H;
    float g[6];
    #pragma unroll
    for (int gi = 0; gi < 3; ++gi) {
      int j = gi * H + hc;
      float a1 = b_ih[j], a2 = b_hh[j];
      const float* wi = w_ih + (size_t)j * H;
      const float* wh = w_hh + (size_t)j * H;
      for (int k = 0; k < H; ++k) {
        a1 = fmaf(cx[k], wi[k], a1);
        a2 = fmaf(st[k], wh[k], a2);
      }
      g[gi] = a1; g[gi + 3] = a2;
    }
    float r  = 1.0f / (1.0f + expf(-(g[0] + g[3])));
    float z  = 1.0f / (1.0f + expf(-(g[1] + g[4])));
    float nn = tanhf(g[2] + r * g[5]);
    out[o] = (1.0f - z) * nn + z * state[o];
  }
}

extern "C" void kernel_launch(void* const* d_in, const int* in_sizes, int n_in,
                              void* d_out, int out_size, void* d_ws, size_t ws_size,
                              hipStream_t stream) {
  const float* state = (const float*)d_in[0];
  const float* nodes = (const float*)d_in[1];
  const float* W1    = (const float*)d_in[2];
  const float* b1    = (const float*)d_in[3];
  const float* W2    = (const float*)d_in[4];
  const float* b2    = (const float*)d_in[5];
  const float* w_ih  = (const float*)d_in[6];
  const float* w_hh  = (const float*)d_in[7];
  const float* b_ih  = (const float*)d_in[8];
  const float* b_hh  = (const float*)d_in[9];
  const int N = in_sizes[1] / H;
  const int nB = (N + TN - 1) / TN;

  char* ws = (char*)d_ws;
  float* hsb = (float*)ws;                    ws += (size_t)BATCH * H * 4;
  unsigned short* Bpk = (unsigned short*)ws;  ws += (size_t)4 * 8 * 32 * 16 * 2;
  float* scores = (float*)ws;                 ws += (size_t)BATCH * N * 4;
  float* pmax = (float*)ws;                   ws += (size_t)nB * BATCH * 4;
  float* gmax = (float*)ws;                   ws += 64;
  float* gsum = (float*)ws;                   ws += 64;
  float* context = (float*)ws;                ws += (size_t)BATCH * H * 4;

  k0_setup<<<1, 256, 0, stream>>>(state, W1, b1, hsb, Bpk, gsum, context);
  k1_scores<<<nB, 256, 0, stream>>>(nodes, Bpk, hsb, W2, b2, scores, pmax, N);
  k2_gmax<<<1, 256, 0, stream>>>(pmax, gmax, nB);
  const int NC = (N + 4095) / 4096;
  k3_sumexp<<<16 * NC, 256, 0, stream>>>(scores, gmax, gsum, N, NC);
  k4_context<<<nB, 256, 0, stream>>>(nodes, scores, gmax, gsum, context, N);
  k5_gru<<<1, 256, 0, stream>>>(context, state, w_ih, w_hh, b_ih, b_hh, (float*)d_out);
}
